// RnnVae_39049842655514
// MI455X (gfx1250) — matrix operand order repaired and checked by reading, not compiled, not run on hardware
//
#include <hip/hip_runtime.h>
#include <math.h>

typedef __attribute__((ext_vector_type(16))) _Float16 v16h;
typedef __attribute__((ext_vector_type(8)))  float    v8f;

#define N_VOCAB  100
#define D_EMB    256
#define D_H      512
#define D_Z      256
#define D_C      128
#define N_LEN    128
#define BSZ      256
#define N_FILT   128
#define D_DEC_IN 640
#define D_DEC_H  384
#define TB       (N_LEN * BSZ)   // 32768 rows (t-major, then b)

static inline int cdiv(int a, int b){ return (a + b - 1) / b; }

// ---------------------------------------------------------------------------
// Generic tiled WMMA GEMM:  C[M,N] = alpha * A[M,K] * Bt[N,K]^T + bias[N]
//   A row-major (f32 or f16, template), Bt row-major N x K (f16), C f32.
//   K must be a multiple of 32 (true for every GEMM in this model).
//   Batched via blockIdx.z with element strides sA/sB/sC.
// Block tile 128x128, 8 waves (256 thr): wave grid 4(M) x 2(N), each wave a
// 32x64 tile = 2x4 WMMA accumulators -> 8 v_wmma per K-step per wave.
// ---------------------------------------------------------------------------
constexpr int BM = 128, BN = 128, BK = 32, LDSW = BK + 2; // padded f16 stride

template <typename AT>
__global__ __launch_bounds__(256)
void gemm_wmma(const AT* __restrict__ A, int lda, long long sA,
               const _Float16* __restrict__ Bt, int ldb, long long sB,
               const float* __restrict__ bias,
               float* __restrict__ C, int ldc, long long sC,
               int M, int N, int K, float alpha)
{
  __shared__ _Float16 As[BM * LDSW];
  __shared__ _Float16 Bs[BN * LDSW];

  const int tid  = threadIdx.x;
  const int lane = tid & 31;
  const int wv   = tid >> 5;
  const int m0   = blockIdx.x * BM;
  const int n0   = blockIdx.y * BN;
  A  += (long long)blockIdx.z * sA;
  Bt += (long long)blockIdx.z * sB;
  C  += (long long)blockIdx.z * sC;

  const int wm = (wv & 3) * 32;   // wave M offset within block tile (0..96)
  const int wn = (wv >> 2) * 64;  // wave N offset within block tile (0/64)

  v8f zero = {};
  v8f acc[2][4] = {{zero, zero, zero, zero}, {zero, zero, zero, zero}};

  // staging: A tile 128x32 and B tile 128x32, 16 f16 per thread each
  const int ar = tid >> 1, ac = (tid & 1) * 16;

  for (int k0 = 0; k0 < K; k0 += BK) {
    __syncthreads();
    { // stage A (convert to f16)
      const int gm = m0 + ar;
      _Float16* dst = &As[ar * LDSW + ac];
      if (gm < M) {
        const AT* ap = A + (long long)gm * lda + k0 + ac;
        #pragma unroll
        for (int i = 0; i < 16; ++i) dst[i] = (_Float16)(float)ap[i];
        if (k0 + BK < K) __builtin_prefetch((const void*)(ap + BK), 0, 1);
      } else {
        #pragma unroll
        for (int i = 0; i < 16; ++i) dst[i] = (_Float16)0.f;
      }
    }
    { // stage Bt
      const int gn = n0 + ar;
      _Float16* dst = &Bs[ar * LDSW + ac];
      if (gn < N) {
        const _Float16* bp = Bt + (long long)gn * ldb + k0 + ac;
        #pragma unroll
        for (int i = 0; i < 16; ++i) dst[i] = bp[i];
        if (k0 + BK < K) __builtin_prefetch((const void*)(bp + BK), 0, 1);
      } else {
        #pragma unroll
        for (int i = 0; i < 16; ++i) dst[i] = (_Float16)0.f;
      }
    }
    __syncthreads();

    // A fragment: lanes 0-15 M rows, K = khA + {0..7} in v0-3, khA+16+{0..7} in v4-7
    v16h afrag[2], bfrag[4];
    const int khA  = (lane >> 4) * 8;
    const int mrow = lane & 15;
    #pragma unroll
    for (int mi = 0; mi < 2; ++mi) {
      const _Float16* src = &As[(wm + mi * 16 + mrow) * LDSW];
      #pragma unroll
      for (int v = 0; v < 4; ++v) {
        afrag[mi][2*v]     = src[khA + 2*v];
        afrag[mi][2*v + 1] = src[khA + 2*v + 1];
        afrag[mi][2*v + 8] = src[khA + 16 + 2*v];
        afrag[mi][2*v + 9] = src[khA + 16 + 2*v + 1];
      }
    }
    // B fragment: K = khB + {0..7} in v0-3, khB+16+{0..7} in v4-7 (col = lane&15)
    const int khB = (lane >> 4) * 8;
    #pragma unroll
    for (int ni = 0; ni < 4; ++ni) {
      const _Float16* src = &Bs[(wn + ni * 16 + (lane & 15)) * LDSW];
      #pragma unroll
      for (int v = 0; v < 8; ++v) {
        bfrag[ni][2*v]     = src[khB + 2*v + (v >= 4 ? 8 : 0)];
        bfrag[ni][2*v + 1] = src[khB + 2*v + 1 + (v >= 4 ? 8 : 0)];
      }
    }
    #pragma unroll
    for (int mi = 0; mi < 2; ++mi)
      #pragma unroll
      for (int ni = 0; ni < 4; ++ni)
        acc[mi][ni] = __builtin_amdgcn_wmma_f32_16x16x32_f16(
            false, afrag[mi], false, bfrag[ni], (short)0, acc[mi][ni],
            false, false);
  }

  // store: C/D layout — lanes 0-15 N cols, VGPR e -> M=e (lo lanes) / e+8 (hi)
  #pragma unroll
  for (int mi = 0; mi < 2; ++mi) {
    #pragma unroll
    for (int ni = 0; ni < 4; ++ni) {
      const int n = n0 + wn + ni * 16 + (lane & 15);
      if (n >= N) continue;
      const float bval = bias ? bias[n] : 0.f;
      #pragma unroll
      for (int e = 0; e < 8; ++e) {
        const int m = m0 + wm + mi * 16 + e + ((lane >> 4) << 3);
        if (m < M) C[(long long)m * ldc + n] = acc[mi][ni][e] * alpha + bval;
      }
    }
  }
}

// ---------------------------------------------------------------------------
// Elementwise / scan kernels
// ---------------------------------------------------------------------------
__global__ void k_transpose_cvt(const float* __restrict__ W, _Float16* __restrict__ Wt,
                                int K, int N) {
  int i = blockIdx.x * blockDim.x + threadIdx.x;
  if (i >= K * N) return;
  int n = i / K, k = i % K;
  Wt[i] = (_Float16)W[k * N + n];
}

__global__ void k_cvt(const float* __restrict__ W, _Float16* __restrict__ Wt, int n) {
  int i = blockIdx.x * blockDim.x + threadIdx.x;
  if (i < n) Wt[i] = (_Float16)W[i];
}

__global__ void k_embed(const int* __restrict__ x, const float* __restrict__ emb,
                        float* __restrict__ xe) {
  int i = blockIdx.x * blockDim.x + threadIdx.x;
  if (i >= TB * D_EMB) return;
  int e = i % D_EMB, rem = i / D_EMB, b = rem % BSZ, t = rem / BSZ;
  xe[i] = emb[x[b * N_LEN + t] * D_EMB + e];
}

// SRU scan: c_t = f*c + (1-f)*x~ ; h_t = r*tanh(c) + (1-r)*xres
// U rows (t*BSZ + b) x 3*dh : [x~ | f | r]
__global__ void k_sru_scan(const float* __restrict__ U, const float* __restrict__ bias,
                           const float* __restrict__ xres, const float* __restrict__ c0,
                           float* __restrict__ h, int dh) {
  int i = blockIdx.x * blockDim.x + threadIdx.x;
  if (i >= BSZ * dh) return;
  int b = i / dh, j = i % dh;
  float c  = c0 ? c0[i] : 0.f;
  float bf = bias[j], br = bias[dh + j];
  for (int t = 0; t < N_LEN; ++t) {
    size_t row = (size_t)t * BSZ + b;
    const float* u = U + row * (size_t)(3 * dh);
    float f = 1.f / (1.f + expf(-(u[dh + j] + bf)));
    float r = 1.f / (1.f + expf(-(u[2 * dh + j] + br)));
    c = f * c + (1.f - f) * u[j];
    h[row * dh + j] = r * tanhf(c) + (1.f - r) * xres[row * dh + j];
  }
}

__global__ void k_z_kl(const float* __restrict__ mu, const float* __restrict__ lv,
                       const float* __restrict__ eps, float* __restrict__ z,
                       float* __restrict__ acc) {
  int i = blockIdx.x * blockDim.x + threadIdx.x;
  if (i >= BSZ * D_Z) return;
  float m = mu[i], l = lv[i];
  z[i] = m + expf(0.5f * l) * eps[i];
  atomicAdd(acc, expf(l) + m * m - 1.f - l);
}

__global__ void k_im2col(const int* __restrict__ x, const float* __restrict__ emb,
                         _Float16* __restrict__ outb, int ksz, int Tout) {
  int i = blockIdx.x * blockDim.x + threadIdx.x;
  int n = BSZ * Tout * D_EMB * ksz;
  if (i >= n) return;
  int j = i % (D_EMB * ksz), row = i / (D_EMB * ksz);
  int c = j / ksz, dk = j % ksz;
  int b = row / Tout, t = row % Tout;
  outb[i] = (_Float16)emb[x[b * N_LEN + t + dk] * D_EMB + c];
}

__global__ void k_maxpool(const float* __restrict__ y, float* __restrict__ feats,
                          int Tout, int foff) {
  int i = blockIdx.x * blockDim.x + threadIdx.x;
  if (i >= BSZ * N_FILT) return;
  int b = i / N_FILT, f = i % N_FILT;
  float m = 0.f;
  for (int t = 0; t < Tout; ++t) {
    float v = y[(size_t)(b * Tout + t) * N_FILT + f];
    v = v > 0.f ? v : 0.f;
    m = v > m ? v : m;
  }
  feats[b * (3 * N_FILT) + foff + f] = m;
}

__global__ void k_softmax_f32(const float* __restrict__ X, float* __restrict__ Y,
                              int rows, int cols) {
  int r = blockIdx.x * blockDim.x + threadIdx.x;
  if (r >= rows) return;
  const float* xr = X + (size_t)r * cols;
  float* yr = Y + (size_t)r * cols;
  float mx = xr[0];
  for (int i = 1; i < cols; ++i) mx = fmaxf(mx, xr[i]);
  float s = 0.f;
  for (int i = 0; i < cols; ++i) s += expf(xr[i] - mx);
  float inv = 1.f / s;
  for (int i = 0; i < cols; ++i) yr[i] = expf(xr[i] - mx) * inv;
}

__global__ void k_softmax_f16(const float* __restrict__ X, _Float16* __restrict__ Y,
                              int rows, int cols) {
  int r = blockIdx.x * blockDim.x + threadIdx.x;
  if (r >= rows) return;
  const float* xr = X + (size_t)r * cols;
  _Float16* yr = Y + (size_t)r * cols;
  float mx = xr[0];
  for (int i = 1; i < cols; ++i) mx = fmaxf(mx, xr[i]);
  float s = 0.f;
  for (int i = 0; i < cols; ++i) s += expf(xr[i] - mx);
  float inv = 1.f / s;
  for (int i = 0; i < cols; ++i) yr[i] = (_Float16)(expf(xr[i] - mx) * inv);
}

__global__ void k_hc(const float* __restrict__ z, const float* __restrict__ c,
                     float* __restrict__ hc) {
  int i = blockIdx.x * blockDim.x + threadIdx.x;
  if (i >= BSZ * D_DEC_H) return;
  int b = i / D_DEC_H, j = i % D_DEC_H;
  hc[i] = (j < D_Z) ? z[b * D_Z + j] : c[b * D_C + (j - D_Z)];
}

__global__ void k_decin(const int* __restrict__ x, const float* __restrict__ drop_u,
                        const float* __restrict__ emb, const float* __restrict__ hc,
                        float* __restrict__ decin) {
  int i = blockIdx.x * blockDim.x + threadIdx.x;
  if (i >= TB * D_DEC_IN) return;
  int j = i % D_DEC_IN, row = i / D_DEC_IN, b = row % BSZ, t = row / BSZ;
  float v;
  if (j < D_EMB) {
    int tok = (drop_u[b * N_LEN + t] < 0.1f) ? 0 : x[b * N_LEN + t];  // UNK=0
    v = emb[tok * D_EMB + j];
  } else {
    v = hc[b * D_DEC_H + (j - D_EMB)];
  }
  decin[i] = v;
}

__global__ void k_obt(const float* __restrict__ o, _Float16* __restrict__ obt,
                      _Float16* __restrict__ obdt) {
  int i = blockIdx.x * blockDim.x + threadIdx.x;
  if (i >= TB * D_DEC_H) return;
  int d = i % D_DEC_H, row = i / D_DEC_H, b = row % BSZ, t = row / BSZ;
  float v = o[i];
  obt [((size_t)b * N_LEN   + t) * D_DEC_H + d] = (_Float16)v;
  obdt[((size_t)b * D_DEC_H + d) * N_LEN   + t] = (_Float16)v;
}

__global__ void k_recon(const float* __restrict__ logits, const int* __restrict__ x,
                        float* __restrict__ acc) {
  int i = blockIdx.x * blockDim.x + threadIdx.x;
  if (i >= BSZ * N_LEN) return;
  int b = i / N_LEN, t = i % N_LEN;
  const float* lg = logits + (size_t)i * N_VOCAB;   // row b*T + t
  float mx = lg[0];
  for (int j = 1; j < N_VOCAB; ++j) mx = fmaxf(mx, lg[j]);
  float s = 0.f;
  for (int j = 0; j < N_VOCAB; ++j) s += expf(lg[j] - mx);
  int tgt = (t < N_LEN - 1) ? x[b * N_LEN + t + 1] : 1;  // PAD=1
  atomicAdd(acc + 1, -(lg[tgt] - mx - logf(s)));
}

__global__ void k_zero(float* acc) { if (threadIdx.x < 2) acc[threadIdx.x] = 0.f; }

__global__ void k_final(const float* __restrict__ acc, float* __restrict__ out) {
  out[0] = 0.5f * acc[0] / (float)BSZ;
  out[1] = acc[1] / (float)(BSZ * N_LEN);
}

// ---------------------------------------------------------------------------
extern "C" void kernel_launch(void* const* d_in, const int* in_sizes, int n_in,
                              void* d_out, int out_size, void* d_ws, size_t ws_size,
                              hipStream_t stream) {
  (void)in_sizes; (void)n_in; (void)out_size; (void)ws_size;
  const int*   x      = (const int*)  d_in[0];
  const float* drop_u = (const float*)d_in[1];
  const float* eps    = (const float*)d_in[2];
  const float* emb    = (const float*)d_in[3];
  const float* enc_W0 = (const float*)d_in[4];
  const float* enc_b0 = (const float*)d_in[5];
  const float* enc_R0 = (const float*)d_in[6];
  const float* enc_W1 = (const float*)d_in[7];
  const float* enc_b1 = (const float*)d_in[8];
  const float* qmu_W  = (const float*)d_in[9];
  const float* qmu_b  = (const float*)d_in[10];
  const float* qlv_W  = (const float*)d_in[11];
  const float* qlv_b  = (const float*)d_in[12];
  const float* dec_W0 = (const float*)d_in[13];
  const float* dec_b0 = (const float*)d_in[14];
  const float* dec_R0 = (const float*)d_in[15];
  const float* dec_W1 = (const float*)d_in[16];
  const float* dec_b1 = (const float*)d_in[17];
  const float* fc_W   = (const float*)d_in[18];
  const float* fc_b   = (const float*)d_in[19];
  const float* c3_W   = (const float*)d_in[20];
  const float* c3_b   = (const float*)d_in[21];
  const float* c4_W   = (const float*)d_in[22];
  const float* c4_b   = (const float*)d_in[23];
  const float* c5_W   = (const float*)d_in[24];
  const float* c5_b   = (const float*)d_in[25];
  const float* disc_W = (const float*)d_in[26];
  const float* disc_b = (const float*)d_in[27];
  float* out = (float*)d_out;

  // ---- workspace bump allocator with phase-based slab reuse ----
  char* ws = (char*)d_ws;
  size_t off = 0;
  auto take = [&](size_t bytes) -> char* {
    char* p = ws + off;
    off += (bytes + 255) & ~(size_t)255;
    return p;
  };
  float* xemb  = (float*)take((size_t)TB * D_EMB * 4);
  char*  Uslab = take((size_t)TB * 1536 * 4);      // enc/dec U; reused: im2col/conv, attention
  float* Ubuf  = (float*)Uslab;
  char*  Rslab = take((size_t)TB * D_H * 4);       // residuals; reused: obt/obdt
  float* xres  = (float*)Rslab;
  float* h0    = (float*)take((size_t)TB * D_H * 4);  // enc h0; reused: dec o0
  float* h1    = (float*)take((size_t)TB * D_H * 4);  // enc h1; reused: dec o1
  char*  Dslab = take((size_t)TB * D_DEC_IN * 4);     // dec_in; reused: logits
  float* decin  = (float*)Dslab;
  float* mu     = (float*)take((size_t)BSZ * D_Z * 4);
  float* logvar = (float*)take((size_t)BSZ * D_Z * 4);
  float* zbuf   = (float*)take((size_t)BSZ * D_Z * 4);
  float* hc     = (float*)take((size_t)BSZ * D_DEC_H * 4);
  float* feats  = (float*)take((size_t)BSZ * 3 * N_FILT * 4);
  float* dlog   = (float*)take((size_t)BSZ * D_C * 4);
  float* ccode  = (float*)take((size_t)BSZ * D_C * 4);
  float* acc    = (float*)take(256);
  _Float16* encW0t = (_Float16*)take((size_t)1536 * 256 * 2);
  _Float16* encR0t = (_Float16*)take((size_t)512  * 256 * 2);
  _Float16* encW1t = (_Float16*)take((size_t)1536 * 512 * 2);
  _Float16* qmuWt  = (_Float16*)take((size_t)256  * 512 * 2);
  _Float16* qlvWt  = (_Float16*)take((size_t)256  * 512 * 2);
  _Float16* decW0t = (_Float16*)take((size_t)1152 * 640 * 2);
  _Float16* decR0t = (_Float16*)take((size_t)384  * 640 * 2);
  _Float16* decW1t = (_Float16*)take((size_t)1152 * 384 * 2);
  _Float16* fcWt   = (_Float16*)take((size_t)100  * 384 * 2);
  _Float16* discWt = (_Float16*)take((size_t)128  * 384 * 2);
  _Float16* c3Wt   = (_Float16*)take((size_t)128  * 768  * 2);
  _Float16* c4Wt   = (_Float16*)take((size_t)128  * 1024 * 2);
  _Float16* c5Wt   = (_Float16*)take((size_t)128  * 1280 * 2);
  // slab aliases (disjoint in time)
  _Float16* im2col = (_Float16*)Uslab;                              // <= 81.3 MB
  float*    convY  = (float*)(Uslab + (size_t)100 * 1024 * 1024);   // 16.6 MB
  float*    scores = (float*)Uslab;                                  // 16.8 MB
  _Float16* a16    = (_Float16*)(Uslab + (size_t)20 * 1024 * 1024); //  8.4 MB
  float*    attn   = (float*)(Uslab + (size_t)32 * 1024 * 1024);    // 50.4 MB
  _Float16* obt    = (_Float16*)Rslab;
  _Float16* obdt   = obt + (size_t)TB * D_DEC_H;
  float*    logits = (float*)Dslab;

  auto gemm_f32 = [&](const float* A, int lda, long long sA,
                      const _Float16* Bt, int ldb, long long sB, const float* bias,
                      float* C, int ldc, long long sC, int M, int N, int K,
                      float alpha, int nb) {
    dim3 g(cdiv(M, BM), cdiv(N, BN), nb);
    gemm_wmma<float><<<g, 256, 0, stream>>>(A, lda, sA, Bt, ldb, sB, bias, C, ldc, sC, M, N, K, alpha);
  };
  auto gemm_f16 = [&](const _Float16* A, int lda, long long sA,
                      const _Float16* Bt, int ldb, long long sB, const float* bias,
                      float* C, int ldc, long long sC, int M, int N, int K,
                      float alpha, int nb) {
    dim3 g(cdiv(M, BM), cdiv(N, BN), nb);
    gemm_wmma<_Float16><<<g, 256, 0, stream>>>(A, lda, sA, Bt, ldb, sB, bias, C, ldc, sC, M, N, K, alpha);
  };
  auto l1d = [&](int n) { return dim3(cdiv(n, 256)); };

  k_zero<<<1, 32, 0, stream>>>(acc);

  // ---- weight prep: transpose+convert to f16 N x K ----
  k_transpose_cvt<<<l1d(256*1536), 256, 0, stream>>>(enc_W0, encW0t, 256, 1536);
  k_transpose_cvt<<<l1d(256*512 ), 256, 0, stream>>>(enc_R0, encR0t, 256, 512);
  k_transpose_cvt<<<l1d(512*1536), 256, 0, stream>>>(enc_W1, encW1t, 512, 1536);
  k_transpose_cvt<<<l1d(512*256 ), 256, 0, stream>>>(qmu_W,  qmuWt,  512, 256);
  k_transpose_cvt<<<l1d(512*256 ), 256, 0, stream>>>(qlv_W,  qlvWt,  512, 256);
  k_transpose_cvt<<<l1d(640*1152), 256, 0, stream>>>(dec_W0, decW0t, 640, 1152);
  k_transpose_cvt<<<l1d(640*384 ), 256, 0, stream>>>(dec_R0, decR0t, 640, 384);
  k_transpose_cvt<<<l1d(384*1152), 256, 0, stream>>>(dec_W1, decW1t, 384, 1152);
  k_transpose_cvt<<<l1d(384*100 ), 256, 0, stream>>>(fc_W,   fcWt,   384, 100);
  k_transpose_cvt<<<l1d(384*128 ), 256, 0, stream>>>(disc_W, discWt, 384, 128);
  k_cvt<<<l1d(128*768 ), 256, 0, stream>>>(c3_W, c3Wt, 128*768);   // (F, C*k) already N x K
  k_cvt<<<l1d(128*1024), 256, 0, stream>>>(c4_W, c4Wt, 128*1024);
  k_cvt<<<l1d(128*1280), 256, 0, stream>>>(c5_W, c5Wt, 128*1280);

  // ---- embedding: (T,B,E) ----
  k_embed<<<l1d(TB*D_EMB), 256, 0, stream>>>(x, emb, xemb);

  // ---- encoder ----
  gemm_f32(xemb, D_EMB, 0, encW0t, D_EMB, 0, nullptr, Ubuf, 3*D_H, 0, TB, 3*D_H, D_EMB, 1.f, 1);
  gemm_f32(xemb, D_EMB, 0, encR0t, D_EMB, 0, nullptr, xres, D_H,   0, TB, D_H,   D_EMB, 1.f, 1);
  k_sru_scan<<<l1d(BSZ*D_H), 256, 0, stream>>>(Ubuf, enc_b0, xres, nullptr, h0, D_H);
  gemm_f32(h0, D_H, 0, encW1t, D_H, 0, nullptr, Ubuf, 3*D_H, 0, TB, 3*D_H, D_H, 1.f, 1);
  k_sru_scan<<<l1d(BSZ*D_H), 256, 0, stream>>>(Ubuf, enc_b1, /*xres=*/h0, nullptr, h1, D_H);

  const float* h_last = h1 + (size_t)(N_LEN - 1) * BSZ * D_H;
  gemm_f32(h_last, D_H, 0, qmuWt, D_H, 0, qmu_b, mu,     D_Z, 0, BSZ, D_Z, D_H, 1.f, 1);
  gemm_f32(h_last, D_H, 0, qlvWt, D_H, 0, qlv_b, logvar, D_Z, 0, BSZ, D_Z, D_H, 1.f, 1);
  k_z_kl<<<l1d(BSZ*D_Z), 256, 0, stream>>>(mu, logvar, eps, zbuf, acc);

  // ---- discriminator (Kim-CNN as im2col GEMMs) ----
  {
    const float* cb[3] = {c3_b, c4_b, c5_b};
    const _Float16* cw[3] = {c3Wt, c4Wt, c5Wt};
    for (int ki = 0; ki < 3; ++ki) {
      int ksz = ki + 3, Tout = N_LEN - ksz + 1, Kc = D_EMB * ksz, Mc = BSZ * Tout;
      k_im2col<<<l1d(Mc*Kc), 256, 0, stream>>>(x, emb, im2col, ksz, Tout);
      gemm_f16(im2col, Kc, 0, cw[ki], Kc, 0, cb[ki], convY, N_FILT, 0, Mc, N_FILT, Kc, 1.f, 1);
      k_maxpool<<<l1d(BSZ*N_FILT), 256, 0, stream>>>(convY, feats, Tout, ki * N_FILT);
    }
  }
  gemm_f32(feats, 3*N_FILT, 0, discWt, 3*N_FILT, 0, disc_b, dlog, D_C, 0, BSZ, D_C, 3*N_FILT, 1.f, 1);
  k_softmax_f32<<<l1d(BSZ), 256, 0, stream>>>(dlog, ccode, BSZ, D_C);
  k_hc<<<l1d(BSZ*D_DEC_H), 256, 0, stream>>>(zbuf, ccode, hc);

  // ---- decoder ----
  k_decin<<<l1d(TB*D_DEC_IN), 256, 0, stream>>>(x, drop_u, emb, hc, decin);
  gemm_f32(decin, D_DEC_IN, 0, decW0t, D_DEC_IN, 0, nullptr, Ubuf, 3*D_DEC_H, 0, TB, 3*D_DEC_H, D_DEC_IN, 1.f, 1);
  gemm_f32(decin, D_DEC_IN, 0, decR0t, D_DEC_IN, 0, nullptr, xres, D_DEC_H,   0, TB, D_DEC_H,   D_DEC_IN, 1.f, 1);
  float* o0 = h0;  // slab reuse
  float* o1 = h1;
  k_sru_scan<<<l1d(BSZ*D_DEC_H), 256, 0, stream>>>(Ubuf, dec_b0, xres, hc, o0, D_DEC_H);
  gemm_f32(o0, D_DEC_H, 0, decW1t, D_DEC_H, 0, nullptr, Ubuf, 3*D_DEC_H, 0, TB, 3*D_DEC_H, D_DEC_H, 1.f, 1);
  k_sru_scan<<<l1d(BSZ*D_DEC_H), 256, 0, stream>>>(Ubuf, dec_b1, /*xres=*/o0, hc, o1, D_DEC_H);

  // ---- self-attention over time (batched WMMA GEMMs) ----
  k_obt<<<l1d(TB*D_DEC_H), 256, 0, stream>>>(o1, obt, obdt);
  const float iscale = 1.f / sqrtf((float)D_DEC_H);
  gemm_f16(obt, D_DEC_H, (long long)N_LEN * D_DEC_H,
           obt, D_DEC_H, (long long)N_LEN * D_DEC_H, nullptr,
           scores, N_LEN, (long long)N_LEN * N_LEN,
           N_LEN, N_LEN, D_DEC_H, iscale, BSZ);
  k_softmax_f16<<<l1d(BSZ*N_LEN), 256, 0, stream>>>(scores, a16, BSZ * N_LEN, N_LEN);
  gemm_f16(a16, N_LEN, (long long)N_LEN * N_LEN,
           obdt, N_LEN, (long long)D_DEC_H * N_LEN, nullptr,
           attn, D_DEC_H, (long long)N_LEN * D_DEC_H,
           N_LEN, D_DEC_H, N_LEN, 1.f, BSZ);

  // ---- output projection + losses ----
  gemm_f32(attn, D_DEC_H, 0, fcWt, D_DEC_H, 0, fc_b, logits, N_VOCAB, 0, TB, N_VOCAB, D_DEC_H, 1.f, 1);
  k_recon<<<l1d(BSZ*N_LEN), 256, 0, stream>>>(logits, x, acc);
  k_final<<<1, 1, 0, stream>>>(acc, out);
}
